// KnotHyperTransformer_16561393893559
// MI455X (gfx1250) — compile-verified, matrix operand order checked by reading, not run on hardware
//
#include <hip/hip_runtime.h>
#include <math.h>

// ---------------- problem constants ----------------
constexpr int Bn    = 16;    // batch
constexpr int Sn    = 2048;  // seq
constexpr int Dn    = 512;   // hidden
constexpr int NHEAD = 4;
constexpr int HDn   = 128;   // head dim
constexpr int Hn    = 683;   // num_hyper = ceil(2048/3)
constexpr int Tn    = 4 + Hn;        // 687 tokens after concat
constexpr int NT    = Bn * Tn;       // 10992 rows

typedef __attribute__((ext_vector_type(16))) __bf16 v16bf;
typedef __attribute__((ext_vector_type(8)))  float  v8f;

union BFrag { unsigned int u[8]; v16bf v; };

__device__ __forceinline__ unsigned short f2bf(float f) {
    unsigned int u = __float_as_uint(f);
    unsigned int r = u + 0x7FFFu + ((u >> 16) & 1u);   // round-to-nearest-even
    return (unsigned short)(r >> 16);
}

// ---------------- fp32 -> bf16 convert ----------------
__global__ void k_cvt(const float* __restrict__ x, unsigned short* __restrict__ y, int n) {
    int i = blockIdx.x * 256 + threadIdx.x;
    if (i < n) y[i] = f2bf(x[i]);
}

// ---------------- chunk means over stride-3 groups ----------------
__global__ void k_chunks(const int* __restrict__ braids, const float* __restrict__ emb,
                         float* __restrict__ chunk) {
    int bh = blockIdx.x;                 // Bn*Hn blocks
    int h = bh % Hn, b = bh / Hn;
    int s0 = h * 3;
    int cnt = (Sn - s0 < 3) ? (Sn - s0) : 3;
    auto ix = [](int v) { v = v < -3 ? -3 : (v > 3 ? 3 : v); return v + 3; };
    int i0 = ix(braids[b * Sn + s0]);
    int i1 = (cnt > 1) ? ix(braids[b * Sn + s0 + 1]) : 0;
    int i2 = (cnt > 2) ? ix(braids[b * Sn + s0 + 2]) : 0;
    float inv = 1.0f / (float)cnt;
    for (int d = threadIdx.x; d < Dn; d += 128) {
        float a = emb[i0 * Dn + d];
        if (cnt > 1) a += emb[i1 * Dn + d];
        if (cnt > 2) a += emb[i2 * Dn + d];
        chunk[(size_t)bh * Dn + d] = a * inv;
    }
}

// ---------------- node rows: mean of all embeddings ----------------
__global__ void k_nodes(const float* __restrict__ chunk, float* __restrict__ comb_f,
                        unsigned short* __restrict__ comb_b) {
    int b = blockIdx.x;
    for (int d = threadIdx.x; d < Dn; d += 256) {
        float s = 0.f;
        for (int h = 0; h < Hn; ++h)
            s += chunk[((size_t)b * Hn + h) * Dn + d] * ((h == Hn - 1) ? 2.f : 3.f);
        float v = s / (float)Sn;
        unsigned short vb = f2bf(v);
        for (int t = 0; t < 4; ++t) {
            comb_f[((size_t)(b * Tn) + t) * Dn + d] = v;
            comb_b[((size_t)(b * Tn) + t) * Dn + d] = vb;
        }
    }
}

// ---------------- hyperedges: framing rotation + flattened roll ----------------
__global__ void k_hyper(const float* __restrict__ chunk, const float* __restrict__ framings,
                        float* __restrict__ comb_f, unsigned short* __restrict__ comb_b) {
    int bh = blockIdx.x;
    int h = bh % Hn, b = bh / Hn;
    float fm = (framings[0] + framings[1] + framings[2]) * (1.f / 3.f);
    float theta = fm * 3.14159265358979f + 0.1f * (float)h;
    float cf = cosf(theta), sf = sinf(theta);
    for (int d = threadIdx.x; d < Dn; d += 128) {
        float cur = chunk[(size_t)bh * Dn + d];
        int flat = b * Dn + d - 1;                 // roll over flattened [b*D]
        if (flat < 0) flat += Bn * Dn;
        int pb = flat / Dn, pd = flat % Dn;
        float prev = chunk[((size_t)pb * Hn + h) * Dn + pd];
        float y = cur * cf - prev * sf;
        comb_f[((size_t)(b * Tn) + 4 + h) * Dn + d] = y;
        comb_b[((size_t)(b * Tn) + 4 + h) * Dn + d] = f2bf(y);
    }
}

// ---------------- WMMA GEMM: C[M,N] = A[M,K] * W[N,K]^T + bias ----------------
// block = 128 threads (4 waves), tile = 64x64, K step 32, bf16 in / f32 acc
__global__ void k_gemm_bf16(const unsigned short* __restrict__ A,
                            const unsigned short* __restrict__ W,
                            const float* __restrict__ bias,
                            float* __restrict__ C, int M, int N, int K) {
    int wave = threadIdx.x >> 5;
    int lane = threadIdx.x & 31;
    int half = lane >> 4;
    int lr   = lane & 15;
    int tile_m = blockIdx.y * 64 + wave * 16;
    int tile_n = blockIdx.x * 64;
    int m = tile_m + lr;
    int ms = (m < M) ? m : (M - 1);           // clamp for loads; store is guarded

    v8f acc[4] = {v8f{}, v8f{}, v8f{}, v8f{}};

    for (int k0 = 0; k0 < K; k0 += 32) {
        BFrag a;
        #pragma unroll
        for (int v = 0; v < 8; ++v) {
            int kk = k0 + ((v < 4) ? (2 * v) : (16 + 2 * (v - 4))) + (half ? 8 : 0);
            a.u[v] = *(const unsigned int*)(A + (size_t)ms * K + kk);
        }
        #pragma unroll
        for (int ns = 0; ns < 4; ++ns) {
            int n = tile_n + ns * 16 + lr;
            BFrag bf;
            #pragma unroll
            for (int v = 0; v < 8; ++v) {
                int kk = k0 + ((v < 4) ? (2 * v) : (16 + 2 * (v - 4))) + (half ? 8 : 0);
                bf.u[v] = *(const unsigned int*)(W + (size_t)n * K + kk);
            }
            acc[ns] = __builtin_amdgcn_wmma_f32_16x16x32_bf16(
                false, a.v, false, bf.v, (short)0, acc[ns], false, false);
        }
    }

    #pragma unroll
    for (int ns = 0; ns < 4; ++ns) {
        int n = tile_n + ns * 16 + lr;
        float bv = (bias && n < N) ? bias[n] : 0.f;
        #pragma unroll
        for (int j = 0; j < 8; ++j) {
            int mm = tile_m + j + (half ? 8 : 0);
            if (mm < M && n < N) C[(size_t)mm * N + n] = acc[ns][j] + bv;
        }
    }
}

// ---------------- attention, hyperedge rows (4 keys each): one wave per row ----
__global__ void k_attn_hyper(const float* __restrict__ qkv, float* __restrict__ ctx) {
    constexpr int GROUPS = (Hn + 3) / 4;       // 171
    int wave = threadIdx.x >> 5, lane = threadIdx.x & 31;
    int gid = blockIdx.x;
    int bh = gid / GROUPS, g = gid % GROUPS;
    int b = bh / NHEAD, hh = bh % NHEAD;
    int hrow = g * 4 + wave;
    if (hrow >= Hn) return;
    int t = 4 + hrow;
    const float* base = qkv + (size_t)(b * Tn) * 3 * Dn;

    float kn[4][4], vn[4][4];
    #pragma unroll
    for (int j = 0; j < 4; ++j)
        #pragma unroll
        for (int i = 0; i < 4; ++i) {
            int d = lane + 32 * i;
            kn[j][i] = base[(size_t)j * 3 * Dn + Dn     + hh * HDn + d];
            vn[j][i] = base[(size_t)j * 3 * Dn + 2 * Dn + hh * HDn + d];
        }
    float s[4] = {0.f, 0.f, 0.f, 0.f};
    #pragma unroll
    for (int i = 0; i < 4; ++i) {
        int d = lane + 32 * i;
        float qd = base[(size_t)t * 3 * Dn + hh * HDn + d];
        #pragma unroll
        for (int j = 0; j < 4; ++j) s[j] += qd * kn[j][i];
    }
    #pragma unroll
    for (int off = 16; off > 0; off >>= 1)
        #pragma unroll
        for (int j = 0; j < 4; ++j) s[j] += __shfl_xor(s[j], off, 32);

    const float scale = 0.0883883476483184f;   // 1/sqrt(128)
    #pragma unroll
    for (int j = 0; j < 4; ++j) s[j] *= scale;
    float mx = fmaxf(fmaxf(s[0], s[1]), fmaxf(s[2], s[3]));
    float p[4], sum = 0.f;
    #pragma unroll
    for (int j = 0; j < 4; ++j) { p[j] = expf(s[j] - mx); sum += p[j]; }
    float inv = 1.f / sum;
    #pragma unroll
    for (int i = 0; i < 4; ++i) {
        int d = lane + 32 * i;
        float c = 0.f;
        #pragma unroll
        for (int j = 0; j < 4; ++j) c += p[j] * vn[j][i];
        ctx[((size_t)(b * Tn + t)) * Dn + hh * HDn + d] = c * inv;
    }
}

// ---------------- attention, node rows (683 keys each): block per (b,head) ----
__global__ void k_attn_node(const float* __restrict__ qkv, float* __restrict__ ctx) {
    __shared__ float qn[4 * HDn];
    __shared__ float p[4 * Hn];
    __shared__ float rmax[4], rinv[4];
    int b = blockIdx.x / NHEAD, hh = blockIdx.x % NHEAD;
    int tid = threadIdx.x;
    const float* base = qkv + (size_t)(b * Tn) * 3 * Dn;

    for (int i = tid; i < 4 * HDn; i += 256) {
        int row = i / HDn, d = i % HDn;
        qn[i] = base[(size_t)row * 3 * Dn + hh * HDn + d];
    }
    __syncthreads();
    const float scale = 0.0883883476483184f;
    for (int key = tid; key < Hn; key += 256) {
        const float* kp = base + (size_t)(4 + key) * 3 * Dn + Dn + hh * HDn;
        float s0 = 0.f, s1 = 0.f, s2 = 0.f, s3 = 0.f;
        for (int d = 0; d < HDn; ++d) {
            float kv = kp[d];
            s0 += qn[d] * kv; s1 += qn[HDn + d] * kv;
            s2 += qn[2 * HDn + d] * kv; s3 += qn[3 * HDn + d] * kv;
        }
        p[0 * Hn + key] = s0 * scale; p[1 * Hn + key] = s1 * scale;
        p[2 * Hn + key] = s2 * scale; p[3 * Hn + key] = s3 * scale;
    }
    __syncthreads();
    if (tid < 4) {
        float mx = -3.4e38f;
        for (int key = 0; key < Hn; ++key) mx = fmaxf(mx, p[tid * Hn + key]);
        rmax[tid] = mx;
    }
    __syncthreads();
    for (int i = tid; i < 4 * Hn; i += 256) p[i] = expf(p[i] - rmax[i / Hn]);
    __syncthreads();
    if (tid < 4) {
        float sm = 0.f;
        for (int key = 0; key < Hn; ++key) sm += p[tid * Hn + key];
        rinv[tid] = 1.f / sm;
    }
    __syncthreads();
    for (int i = tid; i < 4 * HDn; i += 256) {
        int row = i / HDn, d = i % HDn;
        const float* vp = base + 2 * Dn + hh * HDn + d;
        float c = 0.f;
        for (int key = 0; key < Hn; ++key)
            c += p[row * Hn + key] * vp[(size_t)(4 + key) * 3 * Dn];
        ctx[((size_t)(b * Tn + row)) * Dn + hh * HDn + d] = c * rinv[row];
    }
}

// ---------------- rotation-roll along seq + layernorm ----------------
__global__ void k_post(const float* __restrict__ o, const float* __restrict__ rot,
                       const float* __restrict__ g, const float* __restrict__ beta,
                       float* __restrict__ comb_f, unsigned short* __restrict__ comb_b) {
    __shared__ float red[128], red2[128];
    int bt = blockIdx.x;
    int t = bt % Tn, b = bt / Tn;
    float th = (rot[0] + rot[1] + rot[2]) * (1.f / 3.f);
    float ct = cosf(th), st = sinf(th);
    int tp = (t == 0) ? (Tn - 1) : (t - 1);
    const float* orow = o + (size_t)bt * Dn;
    const float* prow = o + (size_t)(b * Tn + tp) * Dn;
    float r[4]; float lsum = 0.f, lsq = 0.f;
    #pragma unroll
    for (int i = 0; i < 4; ++i) {
        int d = threadIdx.x + 128 * i;
        r[i] = orow[d] * ct - prow[d] * st;
        lsum += r[i]; lsq += r[i] * r[i];
    }
    red[threadIdx.x] = lsum; red2[threadIdx.x] = lsq;
    __syncthreads();
    for (int off = 64; off > 0; off >>= 1) {
        if (threadIdx.x < off) {
            red[threadIdx.x]  += red[threadIdx.x + off];
            red2[threadIdx.x] += red2[threadIdx.x + off];
        }
        __syncthreads();
    }
    float mu  = red[0] / (float)Dn;
    float var = red2[0] / (float)Dn - mu * mu;
    float inv = rsqrtf(var + 1e-5f);
    #pragma unroll
    for (int i = 0; i < 4; ++i) {
        int d = threadIdx.x + 128 * i;
        float y = (r[i] - mu) * inv * g[d] + beta[d];
        comb_f[(size_t)bt * Dn + d] = y;
        comb_b[(size_t)bt * Dn + d] = f2bf(y);
    }
}

// ---------------- final seq-mean + MLP head ----------------
__global__ void k_head(const float* __restrict__ comb, const float* __restrict__ w1,
                       const float* __restrict__ b1, const float* __restrict__ w2,
                       const float* __restrict__ b2, float* __restrict__ out) {
    __shared__ float st[Dn];
    __shared__ float hbuf[32];
    int b = blockIdx.x;
    for (int d = threadIdx.x; d < Dn; d += 256) {
        float s = 0.f;
        for (int t = 0; t < Tn; ++t) s += comb[((size_t)(b * Tn) + t) * Dn + d];
        st[d] = s / (float)Tn;
    }
    __syncthreads();
    if (threadIdx.x < 32) {
        float s = b1[threadIdx.x];
        for (int d = 0; d < Dn; ++d) s += st[d] * w1[threadIdx.x * Dn + d];
        hbuf[threadIdx.x] = s > 0.f ? s : 0.f;
    }
    __syncthreads();
    if (threadIdx.x == 0) {
        float o0 = b2[0], o1 = b2[1];
        for (int i = 0; i < 32; ++i) { o0 += hbuf[i] * w2[i]; o1 += hbuf[i] * w2[32 + i]; }
        out[b]      = 1.f / (1.f + expf(-o0));
        out[16 + b] = o1;
    }
}

// ---------------- host launch ----------------
extern "C" void kernel_launch(void* const* d_in, const int* in_sizes, int n_in,
                              void* d_out, int out_size, void* d_ws, size_t ws_size,
                              hipStream_t stream) {
    (void)in_sizes; (void)n_in; (void)out_size; (void)ws_size;
    const int*   braids     = (const int*)d_in[0];
    const float* embedding  = (const float*)d_in[1];
    const float* in_proj_w  = (const float*)d_in[2];
    const float* in_proj_b  = (const float*)d_in[3];
    const float* out_proj_w = (const float*)d_in[4];
    const float* out_proj_b = (const float*)d_in[5];
    const float* rotation   = (const float*)d_in[6];
    const float* ln_g       = (const float*)d_in[7];
    const float* ln_b       = (const float*)d_in[8];
    const float* framings   = (const float*)d_in[9];
    const float* w1         = (const float*)d_in[10];
    const float* b1         = (const float*)d_in[11];
    const float* w2         = (const float*)d_in[12];
    const float* b2         = (const float*)d_in[13];
    float* out = (float*)d_out;

    char* wsp = (char*)d_ws;
    auto carve = [&](size_t bytes) {
        char* p = wsp;
        wsp += (bytes + 255) & ~(size_t)255;
        return p;
    };
    float*          chunk  = (float*)         carve((size_t)Bn * Hn * Dn * 4);
    float*          comb_f = (float*)         carve((size_t)NT * Dn * 4);
    unsigned short* comb_b = (unsigned short*)carve((size_t)NT * Dn * 2);
    float*          qkv    = (float*)         carve((size_t)NT * 3 * Dn * 4);
    float*          ctx    = (float*)         carve((size_t)NT * Dn * 4);
    unsigned short* ctx_b  = (unsigned short*)carve((size_t)NT * Dn * 2);
    float*          oproj  = (float*)         carve((size_t)NT * Dn * 4);
    unsigned short* wq_b   = (unsigned short*)carve((size_t)2 * 3 * Dn * Dn * 2);
    unsigned short* wo_b   = (unsigned short*)carve((size_t)2 * Dn * Dn * 2);

    const int nwq = 2 * 3 * Dn * Dn;   // 2 x 1536 x 512
    const int nwo = 2 * Dn * Dn;       // 2 x 512 x 512
    k_cvt<<<(nwq + 255) / 256, 256, 0, stream>>>(in_proj_w, wq_b, nwq);
    k_cvt<<<(nwo + 255) / 256, 256, 0, stream>>>(out_proj_w, wo_b, nwo);

    k_chunks<<<Bn * Hn, 128, 0, stream>>>(braids, embedding, chunk);
    k_nodes<<<Bn, 256, 0, stream>>>(chunk, comb_f, comb_b);
    k_hyper<<<Bn * Hn, 128, 0, stream>>>(chunk, framings, comb_f, comb_b);

    for (int l = 0; l < 2; ++l) {
        dim3 g1(3 * Dn / 64, (NT + 63) / 64);
        k_gemm_bf16<<<g1, 128, 0, stream>>>(comb_b, wq_b + (size_t)l * 3 * Dn * Dn,
                                            in_proj_b + l * 3 * Dn, qkv, NT, 3 * Dn, Dn);
        k_attn_hyper<<<Bn * NHEAD * ((Hn + 3) / 4), 128, 0, stream>>>(qkv, ctx);
        k_attn_node<<<Bn * NHEAD, 256, 0, stream>>>(qkv, ctx);
        k_cvt<<<(NT * Dn + 255) / 256, 256, 0, stream>>>(ctx, ctx_b, NT * Dn);
        dim3 g2(Dn / 64, (NT + 63) / 64);
        k_gemm_bf16<<<g2, 128, 0, stream>>>(ctx_b, wo_b + (size_t)l * Dn * Dn,
                                            out_proj_b + l * Dn, oproj, NT, Dn, Dn);
        k_post<<<NT, 128, 0, stream>>>(oproj, rotation + l * 3, ln_g + l * Dn,
                                       ln_b + l * Dn, comb_f, comb_b);
    }
    k_head<<<Bn, 256, 0, stream>>>(comb_f, w1, b1, w2, b2, out);
}
